// EncoderAttention_17944373363289
// MI455X (gfx1250) — matrix operand order repaired and checked by reading, not compiled, not run on hardware
//
#include <hip/hip_runtime.h>
#include <hip/hip_bf16.h>

// MI455X / gfx1250 (CDNA5, wave32) implementation.
// B=16384, T=36, F=18, H=128, 3H=384.

typedef __attribute__((ext_vector_type(16))) _Float16 v16h;
typedef __attribute__((ext_vector_type(8)))  float    v8f;

#define B_TOT 16384
#define T_LEN 36
#define F_DIM 18
#define H_DIM 128
#define G3H   384
#define ROWS  64      // batch rows per workgroup
#define NWAVE 4       // 4 waves x 16 rows
#define NTHR  128

// fragment-buffer element counts in workspace (f16)
#define N_WHH (24*4*32*16)   // 49152 : W_hh^T, 24 n-tiles x 4 k-steps
#define N_WIH (24*1*32*16)   // 12288 : W_ih^T (K padded 18->32)
#define N_WH  (3*4*32*16)    //  6144 : Wh^T   (N padded 36->48)
#define N_FRAG_TOT (N_WHH + N_WIH + N_WH)

__device__ inline float fexp2_(float x){ return __builtin_amdgcn_exp2f(x); }
__device__ inline float frcp_(float x){ return __builtin_amdgcn_rcpf(x); }
__device__ inline float ftanh_(float x){
  // tanh(x) = 1 - 2/(exp2(2*log2e*x)+1)
  return 1.f - 2.f * frcp_(fexp2_(x * 2.885390081777927f) + 1.f);
}
__device__ inline float fsigm_(float x){
  return frcp_(1.f + fexp2_(-1.4426950408889634f * x));
}
__device__ inline float fexp_(float x){ return fexp2_(1.4426950408889634f * x); }

__device__ inline v8f wmma16(v16h a, v16h b, v8f c){
  return __builtin_amdgcn_wmma_f32_16x16x32_f16(false, a, false, b, (short)0, c, false, false);
}

union AF { v16h v; uint4 u[2]; };

// ---------------------------------------------------------------------------
// Prep kernel: pre-swizzle constant weights into WMMA B-fragment order (f16).
// B fragment (16x16x32 f16): lane l holds column n = base + (l&15),
// K-range = kstep*32 + (l>>4)*16 + j, j = 0..15, contiguous 16 halves.
// ---------------------------------------------------------------------------
__global__ void prep_frag(const float* __restrict__ Wh,
                          const float* __restrict__ W_ih,
                          const float* __restrict__ W_hh,
                          _Float16* __restrict__ ws) {
  int id = blockIdx.x * blockDim.x + threadIdx.x;
  if (id >= N_FRAG_TOT) return;
  int j    = id & 15;
  int lane = (id >> 4) & 31;
  int frag = id >> 9;
  float val;
  if (frag < 96) {                     // W_hh^T: frag = ntile*4 + kstep
    int nt = frag >> 2, k = frag & 3;
    int kg = k*32 + (lane >> 4)*16 + j;
    int ng = nt*16 + (lane & 15);
    val = W_hh[ng * H_DIM + kg];
  } else if (frag < 120) {             // W_ih^T: K padded 18 -> 32
    int nt = frag - 96;
    int kg = (lane >> 4)*16 + j;
    int ng = nt*16 + (lane & 15);
    val = (kg < F_DIM) ? W_ih[ng * F_DIM + kg] : 0.f;
  } else {                             // Wh^T: N padded 36 -> 48
    int fr = frag - 120;
    int nt = fr >> 2, k = fr & 3;
    int kg = k*32 + (lane >> 4)*16 + j;
    int ng = nt*16 + (lane & 15);
    val = (ng < T_LEN) ? Wh[ng * H_DIM + kg] : 0.f;
  }
  ws[id] = (_Float16)val;
}

// ---------------------------------------------------------------------------
// Main kernel: 64 batch rows per workgroup, 4 waves, T-step recurrent loop.
// Steady-state inner loop touches only LDS (+ the streaming seqs x-row).
// ---------------------------------------------------------------------------
__global__ __launch_bounds__(NTHR, 1)
void encoder_attn_gru(const float* __restrict__ seqs,
                      const float* __restrict__ mask,
                      const float* __restrict__ bh,
                      const float* __restrict__ Ws,
                      const float* __restrict__ bs,
                      const float* __restrict__ Wo,
                      const float* __restrict__ b_ih,
                      const float* __restrict__ b_hh,
                      const _Float16* __restrict__ frag,
                      float* __restrict__ out) {
  // LDS (~260 KB of the 320 KB WGP budget)
  __shared__ __align__(16) _Float16 sWhh[N_WHH];              // 98304 B
  __shared__ __align__(16) _Float16 sWih[N_WIH];              // 24576 B
  __shared__ __align__(16) _Float16 sWh [N_WH];               // 12288 B
  __shared__ __align__(16) _Float16 sLinSig[ROWS*F_DIM*T_LEN];// 82944 B
  __shared__ __align__(16) _Float16 sH[ROWS*H_DIM];           // 16384 B, recurrent state f16
  __shared__ __align__(16) _Float16 sX[ROWS*32];              //  4096 B, att-weighted input, K padded
  __shared__ float sLinH[ROWS*T_LEN];                         //  9216 B
  __shared__ float sMask[ROWS*T_LEN];                         //  9216 B
  __shared__ float sWs[T_LEN*T_LEN];                          //  5184 B
  __shared__ float sBh[T_LEN], sBs[T_LEN], sWo[T_LEN];
  __shared__ float sBih[G3H], sBhh[G3H];

  const int tid    = threadIdx.x;
  const int lane   = tid & 31;
  const int wave   = tid >> 5;          // 0..3
  const int lr     = lane & 15;         // lane's row/col index in a 16-tile
  const int hi     = lane >> 4;         // half selector (A/B fragment striping)
  const int sub    = hi * 16;
  const int rbase  = wave * 16;         // this wave's local row base
  const int wgbase = blockIdx.x * ROWS; // global batch row base

  // ---- stage constants / zero state (one-time) ----
  {
    const uint4* src = (const uint4*)frag;
    uint4* dst = (uint4*)sWhh;
    for (int i = tid; i < N_WHH/8; i += NTHR) dst[i] = src[i];
    src = (const uint4*)(frag + N_WHH);
    dst = (uint4*)sWih;
    for (int i = tid; i < N_WIH/8; i += NTHR) dst[i] = src[i];
    src = (const uint4*)(frag + N_WHH + N_WIH);
    dst = (uint4*)sWh;
    for (int i = tid; i < N_WH/8; i += NTHR) dst[i] = src[i];
  }
  for (int i = tid; i < T_LEN; i += NTHR) { sBh[i] = bh[i]; sBs[i] = bs[i]; sWo[i] = Wo[i]; }
  for (int i = tid; i < T_LEN*T_LEN; i += NTHR) sWs[i] = Ws[i];
  for (int i = tid; i < G3H; i += NTHR) { sBih[i] = b_ih[i]; sBhh[i] = b_hh[i]; }
  for (int i = tid; i < ROWS*T_LEN; i += NTHR) sMask[i] = mask[(size_t)wgbase*T_LEN + i];
  for (int i = tid; i < ROWS*H_DIM; i += NTHR) sH[i] = (_Float16)0.f;
  for (int i = tid; i < ROWS*32;   i += NTHR) sX[i] = (_Float16)0.f;
  __syncthreads();

  // ---- lin_sig[r][f][s] = bs[s] + sum_t seqs[r][t][f] * Ws[s][t]  (constant over steps) ----
  for (int p = tid; p < ROWS*F_DIM; p += NTHR) {
    int r = p / F_DIM, f = p % F_DIM;
    float sv[T_LEN];
    const float* sp = seqs + (size_t)(wgbase + r) * T_LEN * F_DIM + f;
#pragma unroll
    for (int t = 0; t < T_LEN; ++t) sv[t] = sp[t * F_DIM];
    _Float16* dstp = &sLinSig[(r * F_DIM + f) * T_LEN];
    for (int s = 0; s < T_LEN; ++s) {
      float a = sBs[s];
      const float* wr = &sWs[s * T_LEN];
#pragma unroll
      for (int t = 0; t < T_LEN; ++t) a += sv[t] * wr[t];
      dstp[s] = (_Float16)a;
    }
  }
  __syncthreads();

  const v8f zero = {0.f,0.f,0.f,0.f,0.f,0.f,0.f,0.f};
  v8f hC[8], hS[8];
#pragma unroll
  for (int j = 0; j < 8; ++j) { hC[j] = zero; hS[j] = zero; }

  const int hrow = rbase + lr;          // local row this lane stripes for A fragments

  for (int t = 0; t < T_LEN; ++t) {
    // ---- load h A-fragments (16-bit A 16x32 layout) ----
    AF hA[4];
#pragma unroll
    for (int k = 0; k < 4; ++k) {
      const _Float16* p = &sH[hrow * H_DIM + k * 32 + sub];
      hA[k].u[0] = *(const uint4*)p;
      hA[k].u[1] = *(const uint4*)(p + 8);
    }

    // ---- phase 1: lin_h = h @ Wh^T + bh  (N padded to 48) ----
#pragma unroll
    for (int nt = 0; nt < 3; ++nt) {
      v8f acc = zero;
#pragma unroll
      for (int k = 0; k < 4; ++k) {
        AF b;
        const _Float16* bp = &sWh[(size_t)((nt*4 + k)*32 + lane) * 16];
        b.u[0] = *(const uint4*)bp;
        b.u[1] = *(const uint4*)(bp + 8);
        acc = wmma16(hA[k].v, b.v, acc);
      }
      int s = nt*16 + lr;               // C layout: N = lane&15, M = v + 8*(lane>>4)
      if (s < T_LEN) {
        float bb = sBh[s];
#pragma unroll
        for (int v = 0; v < 8; ++v)
          sLinH[(rbase + v + 8*hi) * T_LEN + s] = acc[v] + bb;
      }
    }
    __syncthreads();

    // ---- phase 2: attention (lane = (row, half-of-F)); softmax over F=18 ----
    const int   arow  = rbase + lr;
    const float* lhrow = &sLinH[arow * T_LEN];
    float e[9];
    {
      const _Float16* sig0 = &sLinSig[(arow * F_DIM + hi*9) * T_LEN];
#pragma unroll
      for (int i = 0; i < 9; ++i) e[i] = 0.f;
      for (int s = 0; s < T_LEN; ++s) {
        float lh = lhrow[s];
        float w  = sWo[s];
#pragma unroll
        for (int i = 0; i < 9; ++i)
          e[i] += ftanh_(lh + (float)sig0[i * T_LEN + s]) * w;
      }
    }
    float mx = e[0];
#pragma unroll
    for (int i = 1; i < 9; ++i) mx = fmaxf(mx, e[i]);
    mx = fmaxf(mx, __shfl_xor(mx, 16, 32));
    float sum = 0.f;
#pragma unroll
    for (int i = 0; i < 9; ++i) { e[i] = fexp_(e[i] - mx); sum += e[i]; }
    sum += __shfl_xor(sum, 16, 32);
    float inv = frcp_(sum);             // bo cancels in softmax
    const float* xr = seqs + (size_t)(wgbase + arow) * T_LEN * F_DIM + t * F_DIM;
#pragma unroll
    for (int i = 0; i < 9; ++i) {
      int f = hi*9 + i;
      sX[arow * 32 + f] = (_Float16)(e[i] * inv * xr[f]);
    }
    __syncthreads();

    // ---- phase 3: GRU gates via WMMA, elementwise update in C-layout regs ----
    AF xA;
    {
      const _Float16* p = &sX[hrow * 32 + sub];
      xA.u[0] = *(const uint4*)p;
      xA.u[1] = *(const uint4*)(p + 8);
    }
    float mval[8];
#pragma unroll
    for (int v = 0; v < 8; ++v)
      mval[v] = sMask[(rbase + v + 8*hi) * T_LEN + t];

#pragma unroll
    for (int jt = 0; jt < 8; ++jt) {
      v8f ar = zero, az = zero, ain = zero, ahn = zero;
      AF b;
      // x @ W_ih^T contributions (single K-step, K padded 18->32)
      { const _Float16* bp = &sWih[(size_t)((jt      )*32 + lane) * 16];
        b.u[0] = *(const uint4*)bp; b.u[1] = *(const uint4*)(bp + 8);
        ar = wmma16(xA.v, b.v, ar); }
      { const _Float16* bp = &sWih[(size_t)((8  + jt)*32 + lane) * 16];
        b.u[0] = *(const uint4*)bp; b.u[1] = *(const uint4*)(bp + 8);
        az = wmma16(xA.v, b.v, az); }
      { const _Float16* bp = &sWih[(size_t)((16 + jt)*32 + lane) * 16];
        b.u[0] = *(const uint4*)bp; b.u[1] = *(const uint4*)(bp + 8);
        ain = wmma16(xA.v, b.v, ain); }
      // h @ W_hh^T contributions, 4 K-steps each, fragments from LDS
#pragma unroll
      for (int k = 0; k < 4; ++k) {
        const _Float16* bp;
        bp = &sWhh[(size_t)(((jt      )*4 + k)*32 + lane) * 16];
        b.u[0] = *(const uint4*)bp; b.u[1] = *(const uint4*)(bp + 8);
        ar = wmma16(hA[k].v, b.v, ar);
        bp = &sWhh[(size_t)(((8  + jt)*4 + k)*32 + lane) * 16];
        b.u[0] = *(const uint4*)bp; b.u[1] = *(const uint4*)(bp + 8);
        az = wmma16(hA[k].v, b.v, az);
        bp = &sWhh[(size_t)(((16 + jt)*4 + k)*32 + lane) * 16];
        b.u[0] = *(const uint4*)bp; b.u[1] = *(const uint4*)(bp + 8);
        ahn = wmma16(hA[k].v, b.v, ahn);
      }
      int n = jt*16 + lr;
      float b_r  = sBih[n]       + sBhh[n];
      float b_z  = sBih[128 + n] + sBhh[128 + n];
      float b_in = sBih[256 + n];
      float b_hn = sBhh[256 + n];
#pragma unroll
      for (int v = 0; v < 8; ++v) {
        float rg = fsigm_(ar[v] + b_r);
        float zg = fsigm_(az[v] + b_z);
        float ng = ftanh_(ain[v] + b_in + rg * (ahn[v] + b_hn));
        float hv = (1.f - zg) * ng + zg * hC[jt][v];
        hC[jt][v] = hv;
        hS[jt][v] += mval[v] * hv;
        sH[(rbase + v + 8*hi) * H_DIM + n] = (_Float16)hv;  // state for next step
      }
    }
    __syncthreads();
  }

  // ---- epilogue: latent = sum_t mask*h, from C-layout regs ----
#pragma unroll
  for (int jt = 0; jt < 8; ++jt) {
    int n = jt*16 + lr;
#pragma unroll
    for (int v = 0; v < 8; ++v) {
      int row = wgbase + rbase + v + 8*hi;
      out[(size_t)row * H_DIM + n] = hS[jt][v];
    }
  }
}

// ---------------------------------------------------------------------------
extern "C" void kernel_launch(void* const* d_in, const int* in_sizes, int n_in,
                              void* d_out, int out_size, void* d_ws, size_t ws_size,
                              hipStream_t stream) {
  (void)in_sizes; (void)n_in; (void)out_size; (void)ws_size;
  const float* seqs = (const float*)d_in[0];
  const float* mask = (const float*)d_in[1];
  const float* Wh   = (const float*)d_in[2];
  const float* bh   = (const float*)d_in[3];
  const float* Ws   = (const float*)d_in[4];
  const float* bs   = (const float*)d_in[5];
  const float* Wo   = (const float*)d_in[6];
  // d_in[7] = bo : constant shift, cancels in softmax
  const float* W_ih = (const float*)d_in[8];
  const float* b_ih = (const float*)d_in[9];
  const float* W_hh = (const float*)d_in[10];
  const float* b_hh = (const float*)d_in[11];
  float* out = (float*)d_out;
  _Float16* frag = (_Float16*)d_ws;     // 135,168 bytes of f16 fragments

  prep_frag<<<(N_FRAG_TOT + 255) / 256, 256, 0, stream>>>(Wh, W_ih, W_hh, frag);
  encoder_attn_gru<<<B_TOT / ROWS, NTHR, 0, stream>>>(
      seqs, mask, bh, Ws, bs, Wo, b_ih, b_hh, frag, out);
}
